// VQVAE_39900246180017
// MI455X (gfx1250) — matrix operand order repaired and checked by reading, not compiled, not run on hardware
//
#include <hip/hip_runtime.h>
#include <hip/hip_bf16.h>
#include <math.h>

typedef __attribute__((ext_vector_type(16))) _Float16 v16h;
typedef __attribute__((ext_vector_type(8)))  _Float16 v8h;
typedef __attribute__((ext_vector_type(4)))  _Float16 v4h;
typedef __attribute__((ext_vector_type(8)))  float    v8f;
typedef __attribute__((ext_vector_type(4)))  float    v4f;

union Frag {
    struct { v8h lo, hi; } s;
    v16h v;
};

// ---------------------------------------------------------------------------
// Implicit-GEMM conv (convs, lhs-dilated deconvs, VQ distance GEMM).
// Geometry + tile width are template params: all div/mod fold at compile time.
// Block = 256 threads = 8 wave32s arranged 4(M) x 2(N).
// Block tile 128(M) x BN(N), K-chunk 64, LDS double-buffered.
// BN=128: wave = 32x64 tile -> 8 accums, 16 WMMA / K-iteration.
// BN=64 (narrow layers, e.g. Cout=5): wave = 32x32 -> 8 WMMA / K-iteration.
// Output planes are all powers of two: OW = 1<<ows, OH*OW = 1<<ohws.
// Weights are OIHW: flattened inner offset == k, so B tile = wgt[co*Ktot + k].
// ---------------------------------------------------------------------------
template <int R, int S, int STRIDE, int PAD, int LDIL, int BN>
__global__ __launch_bounds__(256) void conv_wmma_kernel(
    const float* __restrict__ in, const float* __restrict__ wgt,
    const float* __restrict__ bias, const float* __restrict__ residual,
    float* __restrict__ out,
    int N, int Cin, int Hin, int Win,
    int Cout, int ows, int ohws,
    int relu_in, int relu_out)
{
    constexpr int RS = R * S;
    constexpr int NI = BN / 32;           // 16-wide N subtiles per wave

    __shared__ __align__(16) _Float16 lds_a[2][128 * 64];
    __shared__ __align__(16) _Float16 lds_b[2][64 * BN];

    const int tid  = threadIdx.x;
    const int wave = tid >> 5;
    const int lane = tid & 31;
    const int OW   = 1 << ows;
    const int OHW  = 1 << ohws;
    const int Mtot = N << ohws;
    const int Ktot = Cin * RS;
    const int HinWin = Hin * Win;
    const int pixbase  = blockIdx.x * 128;
    const int chanbase = blockIdx.y * BN;
    const int wm = wave >> 1;   // 0..3 : 32-row group
    const int wn = wave & 1;    // 0..1 : (BN/2)-col group

    v8f acc[2][NI] = {};

    // A staging: thread owns 4 consecutive K columns, walks 8 rows (stride 16)
    const int acol0 = (tid & 15) << 2;
    const int arow0 = tid >> 4;
    // B staging: thread owns 4 consecutive N columns
    constexpr int BCOLS   = BN / 4;       // thread-columns
    constexpr int BSTRIDE = 256 / BCOLS;  // K rows per pass
    constexpr int BITER   = 64 / BSTRIDE;
    const int bnc0 = (tid % BCOLS) << 2;
    const int bk0  = tid / BCOLS;

    const int nchunks = (Ktot + 63) >> 6;

    auto stage = [&](int c, int bufsel) {
        const int kk0 = c << 6;
        // ---- A tile: 128 pixels x 64 K (im2col gather, f32 -> f16) ----
        {
            int ci[4], ky[4], kx[4];
            bool kv[4];
#pragma unroll
            for (int j = 0; j < 4; ++j) {
                const int k = kk0 + acol0 + j;
                kv[j] = (k < Ktot);
                ci[j] = k / RS;                    // compile-time divisor
                const int rs = k - ci[j] * RS;
                ky[j] = rs / S;
                kx[j] = rs - ky[j] * S;
            }
            _Float16* dstcol = &lds_a[bufsel][acol0];
#pragma unroll
            for (int it = 0; it < 8; ++it) {
                const int mrow = arow0 + it * 16;
                const int pix  = pixbase + mrow;
                const bool rowok = (pix < Mtot);
                const int n  = pix >> ohws;
                const int t  = pix & (OHW - 1);
                const int oy = t >> ows;
                const int ox = t & (OW - 1);
                const size_t nb = (size_t)n * Cin;
                v4h pack;
#pragma unroll
                for (int j = 0; j < 4; ++j) {
                    float v = 0.0f;
                    if (kv[j] && rowok) {
                        const int ty = oy * STRIDE - PAD + ky[j];
                        const int tx = ox * STRIDE - PAD + kx[j];
                        bool ok = true;
                        int iy = ty, ix = tx;
                        if (LDIL > 1) {            // lhs-dilated (deconv)
                            if ((ty % LDIL) != 0 || (tx % LDIL) != 0) ok = false;
                            iy = ty / LDIL; ix = tx / LDIL;
                        }
                        if (ok && (unsigned)iy < (unsigned)Hin &&
                                  (unsigned)ix < (unsigned)Win) {
                            v = in[(nb + ci[j]) * HinWin + iy * Win + ix];
                            if (relu_in) v = fmaxf(v, 0.0f);
                        }
                    }
                    pack[j] = (_Float16)v;
                }
                *(v4h*)(dstcol + mrow * 64) = pack;   // 8B LDS store
            }
        }
        // ---- B tile: 64 K-rows x BN out-channels (direct OIHW addressing) ----
        {
#pragma unroll
            for (int it = 0; it < BITER; ++it) {
                const int kr = bk0 + it * BSTRIDE;
                const int k  = kk0 + kr;
                const bool kvalid = (k < Ktot);
                v4h pk;
#pragma unroll
                for (int j = 0; j < 4; ++j) {
                    const int co = chanbase + bnc0 + j;
                    float v = (kvalid && co < Cout)
                                  ? wgt[(size_t)co * Ktot + k] : 0.0f;
                    pk[j] = (_Float16)v;
                }
                *(v4h*)&lds_b[bufsel][kr * BN + bnc0] = pk;
            }
            // prefetch weight stream two chunks ahead (global_prefetch_b8)
            const int kp = kk0 + 128 + bk0;
            const int cop = chanbase + bnc0;
            if (kp < Ktot && cop < Cout)
                __builtin_prefetch(&wgt[(size_t)cop * Ktot + kp], 0, 1);
        }
    };

    stage(0, 0);
    __syncthreads();

    for (int c = 0; c < nchunks; ++c) {
        const int cur = c & 1;
        if (c + 1 < nchunks) stage(c + 1, cur ^ 1);   // overlap with compute

        const int sel = lane >> 4;
#pragma unroll
        for (int kc = 0; kc < 2; ++kc) {
            Frag a[2];
#pragma unroll
            for (int mi = 0; mi < 2; ++mi) {
                // A frag: lane = M row (lane&15); K-half select = lane>>4
                const _Float16* abase =
                    &lds_a[cur][(32 * wm + 16 * mi + (lane & 15)) * 64 + 32 * kc];
                a[mi].s.lo = *(const v8h*)(abase + 8 * sel);
                a[mi].s.hi = *(const v8h*)(abase + 16 + 8 * sel);
            }
#pragma unroll
            for (int ni = 0; ni < NI; ++ni) {
                // B frag: lane = N col (lane&15); 16 K values per lane, same k order as A
                Frag b;
                const _Float16* bbase =
                    &lds_b[cur][(32 * kc + 8 * sel) * BN + (BN / 2) * wn + 16 * ni + (lane & 15)];
#pragma unroll
                for (int j = 0; j < 8; ++j) {
                    b.v[j]     = bbase[j * BN];
                    b.v[8 + j] = bbase[(16 + j) * BN];
                }
#pragma unroll
                for (int mi = 0; mi < 2; ++mi)
                    acc[mi][ni] = __builtin_amdgcn_wmma_f32_16x16x32_f16(
                        false, a[mi].v, false, b.v, (short)0, acc[mi][ni], false, false);
            }
        }
        __syncthreads();
    }

    // ---- epilogue: D layout — M = vgpr + 8*(lane>>4), N = lane&15 ----
    // Rows r map to 8 consecutive output pixels per lane -> 16B stores.
    const int lrow = lane >> 4;
    const int lcol = lane & 15;
#pragma unroll
    for (int mi = 0; mi < 2; ++mi) {
        const int pix0 = pixbase + 32 * wm + 16 * mi + 8 * lrow;
        if (pix0 >= Mtot) continue;
        const int n  = pix0 >> ohws;
        const int t0 = pix0 & (OHW - 1);
#pragma unroll
        for (int ni = 0; ni < NI; ++ni) {
            const int co = chanbase + (BN / 2) * wn + 16 * ni + lcol;
            if (co >= Cout) continue;
            const float bv = bias ? bias[co] : 0.0f;
            const size_t off = (((size_t)n * Cout + co) << ohws) + t0;
            v4f q0, q1;
#pragma unroll
            for (int r = 0; r < 4; ++r) {
                float v0 = acc[mi][ni][r] + bv;
                float v1 = acc[mi][ni][r + 4] + bv;
                if (relu_out) { v0 = fmaxf(v0, 0.0f); v1 = fmaxf(v1, 0.0f); }
                q0[r] = v0; q1[r] = v1;
            }
            if (residual) {
                q0 += *(const v4f*)(residual + off);
                q1 += *(const v4f*)(residual + off + 4);
            }
            *(v4f*)(out + off)     = q0;
            *(v4f*)(out + off + 4) = q1;
        }
    }
}

// ---------------------------------------------------------------------------
__global__ void bn_kernel(float* __restrict__ z, const float* __restrict__ g,
                          const float* __restrict__ be, const float* __restrict__ mu,
                          const float* __restrict__ var, int C, int HW, int total)
{
    int i = blockIdx.x * blockDim.x + threadIdx.x;
    if (i >= total) return;
    int c = (i / HW) % C;
    float s = g[c] * rsqrtf(var[c] + 1e-5f);
    z[i] = (z[i] - mu[c]) * s + be[c];
}

__global__ void cnorm_kernel(const float* __restrict__ cb, float* __restrict__ cn,
                             int K, int D)
{
    int k = blockIdx.x * blockDim.x + threadIdx.x;
    if (k >= K) return;
    float s = 0.0f;
    for (int d = 0; d < D; ++d) { float v = cb[(size_t)k * D + d]; s += v * v; }
    cn[k] = s;
}

// argmin over (||c||^2 - 2 z.c): one wave32 per row
__global__ void argmin_kernel(const float* __restrict__ scores,
                              const float* __restrict__ cn,
                              int* __restrict__ idx, int M, int K, int HW)
{
    const int wave = threadIdx.x >> 5, lane = threadIdx.x & 31;
    const int m = blockIdx.x * 8 + wave;
    if (m >= M) return;
    const int n = m / HW, t = m - (m / HW) * HW;
    float best = 3.0e38f;
    int bi = 0;
    for (int co = lane; co < K; co += 32) {
        float s = cn[co] - 2.0f * scores[((size_t)n * K + co) * HW + t];
        if (s < best) { best = s; bi = co; }
    }
#pragma unroll
    for (int off = 16; off > 0; off >>= 1) {
        float ob = __shfl_down(best, off, 32);
        int   oi = __shfl_down(bi, off, 32);
        if (ob < best || (ob == best && oi < bi)) { best = ob; bi = oi; }
    }
    if (lane == 0) idx[m] = bi;
}

// gather q (forward value of straight-through output), commitment + histogram
__global__ void vq_gather_kernel(const float* __restrict__ z,
                                 const float* __restrict__ cb,
                                 const int* __restrict__ idx,
                                 float* __restrict__ zq,
                                 float* __restrict__ commit_sum,
                                 int* __restrict__ hist,
                                 int M, int D, int HW)
{
    int i = blockIdx.x * blockDim.x + threadIdx.x;
    float local = 0.0f;
    if (i < M * D) {
        int m = i / D, d = i - (i / D) * D;
        int n = m / HW, t = m - (m / HW) * HW;
        int k = idx[m];
        float q = cb[(size_t)k * D + d];
        size_t zoff = ((size_t)n * D + d) * HW + t;
        float diff = z[zoff] - q;
        local = diff * diff;
        zq[zoff] = q;
        if (d == 0) atomicAdd(&hist[k], 1);
    }
#pragma unroll
    for (int off = 16; off > 0; off >>= 1) local += __shfl_down(local, off, 32);
    if ((threadIdx.x & 31) == 0) atomicAdd(commit_sum, local);
}

__global__ void zero_kernel(int* __restrict__ hist, float* __restrict__ commit, int n)
{
    int i = blockIdx.x * blockDim.x + threadIdx.x;
    if (i < n) hist[i] = 0;
    if (i == 0) *commit = 0.0f;
}

__global__ void finalize_kernel(const int* __restrict__ hist,
                                const float* __restrict__ commit,
                                float* __restrict__ out2, int Kc, int M, int MD)
{
    __shared__ float sh[512];
    int t = threadIdx.x;
    float p = (t < Kc) ? ((float)hist[t] / (float)M) : 0.0f;
    sh[t] = -p * logf(p + 1e-10f);
    __syncthreads();
    for (int s = 256; s > 0; s >>= 1) {
        if (t < s) sh[t] += sh[t + s];
        __syncthreads();
    }
    if (t == 0) {
        out2[0] = 0.25f * ((*commit) / (float)MD);  // vq_loss
        out2[1] = expf(sh[0]);                      // perplexity
    }
}

// ---------------------------------------------------------------------------
extern "C" void kernel_launch(void* const* d_in, const int* in_sizes, int n_in,
                              void* d_out, int out_size, void* d_ws, size_t ws_size,
                              hipStream_t stream)
{
    (void)in_sizes; (void)n_in; (void)out_size; (void)ws_size;

    const float* x       = (const float*)d_in[0];
    const float* e1_w    = (const float*)d_in[1];
    const float* e1_b    = (const float*)d_in[2];
    const float* e2_w    = (const float*)d_in[3];
    const float* e2_b    = (const float*)d_in[4];
    const float* e3_w    = (const float*)d_in[5];
    const float* e3_b    = (const float*)d_in[6];
    const float* enc_rw1 = (const float*)d_in[7];
    const float* enc_rb1 = (const float*)d_in[8];
    const float* enc_rw2 = (const float*)d_in[9];
    const float* enc_rb2 = (const float*)d_in[10];
    const float* lat_w   = (const float*)d_in[11];
    const float* lat_b   = (const float*)d_in[12];
    const float* bn_g    = (const float*)d_in[13];
    const float* bn_be   = (const float*)d_in[14];
    const float* bn_mu   = (const float*)d_in[15];
    const float* bn_var  = (const float*)d_in[16];
    const float* cb      = (const float*)d_in[17];
    const float* din_w   = (const float*)d_in[18];
    const float* din_b   = (const float*)d_in[19];
    const float* dec_rw1 = (const float*)d_in[20];
    const float* dec_rb1 = (const float*)d_in[21];
    const float* dec_rw2 = (const float*)d_in[22];
    const float* dec_rb2 = (const float*)d_in[23];
    const float* d1_w    = (const float*)d_in[24];
    const float* d1_b    = (const float*)d_in[25];
    const float* d2_w    = (const float*)d_in[26];
    const float* d2_b    = (const float*)d_in[27];
    const float* d3_w    = (const float*)d_in[28];
    const float* d3_b    = (const float*)d_in[29];

    float* outp = (float*)d_out;

    // workspace carve-up (element offsets)
    float* P   = (float*)d_ws;             // 33,554,432 f (134 MB)  ping
    float* Q   = P + 33554432ull;          // 16,777,216 f ( 67 MB)  pong
    float* Sb  = Q + 16777216ull;          //  4,194,304 f ( 17 MB)  small temp
    float* cn  = Sb + 4194304ull;          //        512 f  codebook norms
    int*   idx = (int*)(cn + 512);         //     16,384 i
    int*   hist = idx + 16384;             //        512 i
    float* commit = (float*)(hist + 512);  //          1 f

    // kind: 0 = k4 s2 p1, 1 = k3 s1 p1, 2 = 1x1, 3 = deconv, 4 = deconv narrow
    auto conv = [&](int kind, const float* in, const float* w, const float* b,
                    const float* res, float* out, int N, int Cin, int Hin, int Win,
                    int Cout, int ows, int ohws, int ri, int ro) {
        const int bn = (kind == 4) ? 64 : 128;
        dim3 grid((unsigned)(((N << ohws) + 127) / 128),
                  (unsigned)((Cout + bn - 1) / bn));
        switch (kind) {
        case 0: conv_wmma_kernel<4, 4, 2, 1, 1, 128><<<grid, 256, 0, stream>>>(
                    in, w, b, res, out, N, Cin, Hin, Win, Cout, ows, ohws, ri, ro);
                break;
        case 1: conv_wmma_kernel<3, 3, 1, 1, 1, 128><<<grid, 256, 0, stream>>>(
                    in, w, b, res, out, N, Cin, Hin, Win, Cout, ows, ohws, ri, ro);
                break;
        case 2: conv_wmma_kernel<1, 1, 1, 0, 1, 128><<<grid, 256, 0, stream>>>(
                    in, w, b, res, out, N, Cin, Hin, Win, Cout, ows, ohws, ri, ro);
                break;
        case 3: conv_wmma_kernel<4, 4, 1, 2, 2, 128><<<grid, 256, 0, stream>>>(
                    in, w, b, res, out, N, Cin, Hin, Win, Cout, ows, ohws, ri, ro);
                break;
        default: conv_wmma_kernel<4, 4, 1, 2, 2, 64><<<grid, 256, 0, stream>>>(
                    in, w, b, res, out, N, Cin, Hin, Win, Cout, ows, ohws, ri, ro);
                break;
        }
    };

    const int Nb = 16, hc = 128, D = 128, Kc = 512;

    // ---------------- encoder ----------------
    conv(0, x, e1_w, e1_b, nullptr, P, Nb, 3,   256, 256, hc,     7, 14, 0, 1);
    conv(0, P, e2_w, e2_b, nullptr, Q, Nb, hc,  128, 128, hc,     6, 12, 0, 1);
    conv(0, Q, e3_w, e3_b, nullptr, P, Nb, hc,   64,  64, 2 * hc, 5, 10, 0, 1);

    float* cur = P; float* nxt = Q;
    for (int i = 0; i < 4; ++i) {
        const float* w1 = enc_rw1 + (size_t)i * 256 * 256 * 9;
        const float* b1 = enc_rb1 + (size_t)i * 256;
        const float* w2 = enc_rw2 + (size_t)i * 256 * 256 * 9;
        const float* b2 = enc_rb2 + (size_t)i * 256;
        conv(1, cur, w1, b1, nullptr, Sb, Nb, 256, 32, 32, 256, 5, 10, 1, 0);
        conv(1, Sb,  w2, b2, cur,     nxt, Nb, 256, 32, 32, 256, 5, 10, 1, 0);
        float* tmp = cur; cur = nxt; nxt = tmp;
    }
    // cur == P

    conv(2, cur, lat_w, lat_b, nullptr, Q, Nb, 256, 32, 32, D, 5, 10, 0, 0);
    bn_kernel<<<(Nb * D * 1024 + 255) / 256, 256, 0, stream>>>(Q, bn_g, bn_be, bn_mu, bn_var,
                                                               D, 1024, Nb * D * 1024);

    // ---------------- VQ ----------------
    zero_kernel<<<2, 256, 0, stream>>>(hist, commit, Kc);
    // scores[n][k][pix] = z . codebook_k  (codebook as 1x1 "weights")
    conv(2, Q, cb, nullptr, nullptr, P, Nb, D, 32, 32, Kc, 5, 10, 0, 0);
    cnorm_kernel<<<2, 256, 0, stream>>>(cb, cn, Kc, D);
    argmin_kernel<<<16384 / 8, 256, 0, stream>>>(P, cn, idx, 16384, Kc, 1024);
    vq_gather_kernel<<<(16384 * D + 255) / 256, 256, 0, stream>>>(Q, cb, idx, Sb, commit, hist,
                                                                  16384, D, 1024);

    // ---------------- decoder ----------------
    conv(2, Sb, din_w, din_b, nullptr, P, Nb, D, 32, 32, 256, 5, 10, 0, 0);

    cur = P; nxt = Q;
    for (int i = 0; i < 4; ++i) {
        const float* w1 = dec_rw1 + (size_t)i * 256 * 256 * 9;
        const float* b1 = dec_rb1 + (size_t)i * 256;
        const float* w2 = dec_rw2 + (size_t)i * 256 * 256 * 9;
        const float* b2 = dec_rb2 + (size_t)i * 256;
        conv(1, cur, w1, b1, nullptr, Sb, Nb, 256, 32, 32, 256, 5, 10, 1, 0);
        conv(1, Sb,  w2, b2, cur,     nxt, Nb, 256, 32, 32, 256, 5, 10, 1, 0);
        float* tmp = cur; cur = nxt; nxt = tmp;
    }
    // cur == P

    // deconvs as lhs-dilated convs (ldil=2, pad=2, stride=1, weights pre-flipped OIHW)
    conv(3, cur, d1_w, d1_b, nullptr, Q,    Nb, 256,  32,  32, 256, 6, 12, 0, 1);
    conv(3, Q,   d2_w, d2_b, nullptr, P,    Nb, 256,  64,  64, 128, 7, 14, 0, 1);
    conv(4, P,   d3_w, d3_b, nullptr, outp, Nb, 128, 128, 128,   5, 8, 16, 0, 0);

    // scalars appended after recon (16*5*256*256 = 5,242,880 floats)
    finalize_kernel<<<1, 512, 0, stream>>>(hist, commit, outp + 5242880ull, Kc, 16384, 16384 * D);
}
